// SymAttention_12103217840516
// MI455X (gfx1250) — matrix operand order repaired and checked by reading, not compiled, not run on hardware
//
#include <hip/hip_runtime.h>
#include <hip/hip_bf16.h>
#include <math.h>

typedef __attribute__((ext_vector_type(16))) _Float16 v16h;
typedef __attribute__((ext_vector_type(8)))  _Float16 v8h;
typedef __attribute__((ext_vector_type(8)))  float    v8f;
typedef __attribute__((ext_vector_type(4)))  unsigned int v4u;
typedef __attribute__((ext_vector_type(8)))  int v8i;
typedef __attribute__((ext_vector_type(4)))  int v4i;

// Problem constants
#define NB    64      // batch
#define NY    32
#define NX    32
#define NF    10      // flavours
#define NQKV  1536
#define NVDIM 512     // attention output width
#define KQKV  576     // 512 (cells) + 64 (mean_F fcells)
#define MCELL 65536   // NB*NY*NX
#define LDSS  40      // padded LDS row stride in halfs (80B = 16 DW data + 4 DW pad)

// Tensor Data Mover path (gfx1250 TDM). Builtin arity differs between toolchains:
// ROCm 7.2 (clang-22): 5 args; amdgpu-toolchain (clang-23): 6 args.
#if defined(__has_builtin)
#if __has_builtin(__builtin_amdgcn_tensor_load_to_lds) && \
    __has_builtin(__builtin_amdgcn_s_wait_tensorcnt)
#define USE_TDM 1
#endif
#endif
#ifndef USE_TDM
#define USE_TDM 0
#endif

#if USE_TDM
// Issue one TDM 2D tile load: 128 rows x 32 halfs from row-major global
// (row stride = stride_elems halfs) into LDS at lds_off with the padded
// 40-half row stride (pad_interval = 16 DWORDs, pad_amount = 4 DWORDs).
// Descriptor layout per CDNA5 ISA ch.8 (D# groups 0/1; groups 2/3 unused for 2D).
__device__ __forceinline__ void tdm_load_tile(const _Float16* gsrc,
                                              unsigned lds_off,
                                              int stride_elems) {
  const unsigned long long ga = (unsigned long long)(size_t)gsrc;
  v4u g0;
  g0[0] = 1u;                                       // count=1, user descriptor
  g0[1] = lds_off;                                  // lds_addr (bytes)
  g0[2] = (unsigned)(ga & 0xffffffffu);             // global_addr[31:0]
  g0[3] = (unsigned)((ga >> 32) & 0x1ffffffu)       // global_addr[56:32]
        | (2u << 30);                               // type=2 ("image")
  const unsigned w0 = (1u << 16)                    // data_size = 2 bytes
                    | (1u << 20)                    // pad_enable
                    | (3u << 22)                    // pad_interval: 16 DWORDs
                    | (3u << 25);                   // pad_amount: 4 DWORDs
  const unsigned td0 = (unsigned)stride_elems;      // tensor_dim0 (elems)
  const unsigned td1 = 1u << 20;                    // tensor_dim1 (rows, generous)
  const unsigned tile0 = 32u, tile1 = 128u;
  v8i g1;
  g1[0] = (int)w0;
  g1[1] = (int)((td0 & 0xffffu) << 16);             // [47:32]=abar_addr=0, [63:48]=td0.lo
  g1[2] = (int)(((td0 >> 16) & 0xffffu) | ((td1 & 0xffffu) << 16));
  g1[3] = (int)(((td1 >> 16) & 0xffffu) | (tile0 << 16));   // tile_dim0
  g1[4] = (int)tile1;                               // tile_dim1; tile_dim2=0
  g1[5] = (int)td0;                                 // tensor_dim0_stride[31:0]
  g1[6] = 0;                                        // stride0.hi, stride1.lo
  g1[7] = 0;                                        // stride1.hi
  const v4i z4 = {0, 0, 0, 0};
#if defined(__clang_major__) && (__clang_major__ >= 23)
  const v8i z8 = {0, 0, 0, 0, 0, 0, 0, 0};
  __builtin_amdgcn_tensor_load_to_lds(g0, g1, z4, z4, z8, 0);
#else
  __builtin_amdgcn_tensor_load_to_lds(g0, g1, z4, z4, 0);
#endif
}
#endif // USE_TDM

// ======================= prep kernels =======================

// A matrix for qkv GEMM: [65536, 576] f16 = [cells | mean_F(fcells)]
__global__ void prepA_kernel(const float* __restrict__ cells,
                             const float* __restrict__ fcells,
                             _Float16* __restrict__ Aq) {
  const int m = blockIdx.x;
  for (int k = threadIdx.x; k < KQKV; k += blockDim.x) {
    float v;
    if (k < 512) {
      v = cells[(size_t)m * 512 + k];
    } else {
      const int c = k - 512;
      float s = 0.f;
      #pragma unroll
      for (int f = 0; f < NF; ++f) s += fcells[((size_t)m * NF + f) * 64 + c];
      v = s * (1.0f / NF);
    }
    Aq[(size_t)m * KQKV + k] = (_Float16)v;
  }
}

// W^T for qkv GEMM: [1536][576] f16, rows 0..511 = W_cells, 512..575 = W_fcells[64:]
__global__ void prepWqkvT_kernel(const float* __restrict__ W_cells,
                                 const float* __restrict__ W_fcells,
                                 _Float16* __restrict__ Wt) {
  const int n = blockIdx.x;
  for (int k = threadIdx.x; k < KQKV; k += blockDim.x) {
    float v = (k < 512) ? W_cells[(size_t)k * NQKV + n]
                        : W_fcells[(size_t)(64 + k - 512) * NQKV + n];
    Wt[(size_t)n * KQKV + k] = (_Float16)v;
  }
}

// W_out_cells^T: [512][512] f16
__global__ void prepWoutT_kernel(const float* __restrict__ W, _Float16* __restrict__ Wt) {
  const int n = blockIdx.x;
  for (int k = threadIdx.x; k < 512; k += blockDim.x)
    Wt[(size_t)n * 512 + k] = (_Float16)W[(size_t)k * 512 + n];
}

// per-batch bias: bias[b,n] = globl@W_globl + mean_F(flavours)@(W_fcells[:64] + W_flavours)
__global__ void bias_kernel(const float* __restrict__ globl,
                            const float* __restrict__ flavours,
                            const float* __restrict__ W_globl,
                            const float* __restrict__ W_fcells,
                            const float* __restrict__ W_flavours,
                            float* __restrict__ bias) {
  const int b = blockIdx.x;
  __shared__ float flm[64];
  __shared__ float gl[256];
  const int tid = threadIdx.x;
  if (tid < 64) {
    float s = 0.f;
    #pragma unroll
    for (int f = 0; f < NF; ++f) s += flavours[((size_t)b * NF + f) * 64 + tid];
    flm[tid] = s * (1.0f / NF);
  }
  gl[tid] = globl[(size_t)b * 256 + tid];
  __syncthreads();
  for (int n = tid; n < NQKV; n += blockDim.x) {
    float acc = 0.f;
    for (int c = 0; c < 256; ++c) acc += gl[c] * W_globl[(size_t)c * NQKV + n];
    for (int c = 0; c < 64; ++c)
      acc += flm[c] * (W_fcells[(size_t)c * NQKV + n] + W_flavours[(size_t)c * NQKV + n]);
    bias[(size_t)b * NQKV + n] = acc;
  }
}

// dst[m,n] = src[m, 0:256] @ W[256,1536]; grid (6, M), used for rows and cols biases
__global__ void dense256_kernel(const float* __restrict__ src,
                                const float* __restrict__ W,
                                float* __restrict__ dst) {
  const int m = blockIdx.y;
  const int n = blockIdx.x * 256 + threadIdx.x;
  __shared__ float sr[256];
  sr[threadIdx.x] = src[(size_t)m * 256 + threadIdx.x];
  __syncthreads();
  float acc = 0.f;
  for (int c = 0; c < 256; ++c) acc += sr[c] * W[(size_t)c * NQKV + n];
  dst[(size_t)m * NQKV + n] = acc;
}

// ======================= WMMA GEMM core =======================
// 128x128 block tile, 8 waves (wave32), each wave = 32(M)x64(N) = 2x4 wmma tiles.
// A: [M,KSZ] f16 row-major; Bt: [N,KSZ] f16 (i.e. B transposed). C = A @ Bt^T.
template <int KSZ>
__device__ __forceinline__ void gemm_core(const _Float16* __restrict__ A,
                                          const _Float16* __restrict__ Bt,
                                          int m0, int n0,
                                          _Float16* __restrict__ As,
                                          _Float16* __restrict__ Bs,
                                          v8f c[2][4]) {
  const int tid  = threadIdx.x;
  const int lane = tid & 31;
  const int wave = tid >> 5;
  const int wm = wave & 3;        // 4 M-subtiles of 32
  const int wn = wave >> 2;       // 2 N-subtiles of 64
  const int lrow = lane & 15;
  const int lhi  = lane >> 4;     // lane half

  #pragma unroll
  for (int mi = 0; mi < 2; ++mi)
    #pragma unroll
    for (int ni = 0; ni < 4; ++ni)
      #pragma unroll
      for (int e = 0; e < 8; ++e) c[mi][ni][e] = 0.f;

  for (int k0 = 0; k0 < KSZ; k0 += 32) {
#if USE_TDM
    // One wave drives the Tensor Data Mover for both 128x32 tiles; the
    // padded-LDS layout is produced by the TDM pad fields. TENSORcnt waited
    // in the issuing wave, then the workgroup barrier publishes LDS.
    if (tid < 32) {
      tdm_load_tile(A + (size_t)m0 * KSZ + k0, (unsigned)(size_t)As, KSZ);
      tdm_load_tile(Bt + (size_t)n0 * KSZ + k0, (unsigned)(size_t)Bs, KSZ);
      __builtin_amdgcn_s_wait_tensorcnt(0);
    }
#else
    // cooperative staging: 2 x uint4 per thread per tile
    #pragma unroll
    for (int j = 0; j < 2; ++j) {
      const int u  = tid + 256 * j;     // 0..511
      const int r  = u >> 2;            // row 0..127
      const int c4 = u & 3;             // 16B chunk within 32-half row
      *(uint4*)(As + r * LDSS + c4 * 8) =
          *(const uint4*)(A + (size_t)(m0 + r) * KSZ + k0 + c4 * 8);
      *(uint4*)(Bs + r * LDSS + c4 * 8) =
          *(const uint4*)(Bt + (size_t)(n0 + r) * KSZ + k0 + c4 * 8);
    }
#endif
    __syncthreads();
    if (k0 + 32 < KSZ) {  // prefetch next K tile (global_prefetch_b8)
      __builtin_prefetch(A + (size_t)(m0 + (tid >> 1)) * KSZ + k0 + 32, 0, 0);
      __builtin_prefetch(Bt + (size_t)(n0 + (tid >> 1)) * KSZ + k0 + 32, 0, 0);
    }

    // fragments (ISA 16-bit A 16x32 layout: lane half selects K 0-7/16-23 vs 8-15/24-31;
    // B 32x16: lane = N col, same K split per lane half as A)
    v16h a[2], b[4];
    #pragma unroll
    for (int mi = 0; mi < 2; ++mi) {
      const int row = wm * 32 + mi * 16 + lrow;
      v8h lo = *(const v8h*)(As + row * LDSS + lhi * 8);
      v8h hi = *(const v8h*)(As + row * LDSS + 16 + lhi * 8);
      a[mi] = __builtin_shufflevector(lo, hi, 0,1,2,3,4,5,6,7,8,9,10,11,12,13,14,15);
    }
    #pragma unroll
    for (int ni = 0; ni < 4; ++ni) {
      const int row = wn * 64 + ni * 16 + lrow;
      v8h lo = *(const v8h*)(Bs + row * LDSS + lhi * 8);
      v8h hi = *(const v8h*)(Bs + row * LDSS + 16 + lhi * 8);
      b[ni] = __builtin_shufflevector(lo, hi, 0,1,2,3,4,5,6,7,8,9,10,11,12,13,14,15);
    }

    #pragma unroll
    for (int mi = 0; mi < 2; ++mi)
      #pragma unroll
      for (int ni = 0; ni < 4; ++ni)
        c[mi][ni] = __builtin_amdgcn_wmma_f32_16x16x32_f16(
            false, a[mi], false, b[ni], (short)0, c[mi][ni], false, false);
    __syncthreads();
  }
}

// qkv GEMM + 3-way bias epilogue. grid (N/128=12, M/128=512)
__global__ void __launch_bounds__(256)
gemm_qkv_kernel(const _Float16* __restrict__ A, const _Float16* __restrict__ Bt,
                const float* __restrict__ biasb, const float* __restrict__ rowb,
                const float* __restrict__ colb, float* __restrict__ qkv) {
  __shared__ __attribute__((aligned(16))) _Float16 As[128 * LDSS];
  __shared__ __attribute__((aligned(16))) _Float16 Bs[128 * LDSS];
  const int m0 = blockIdx.y * 128, n0 = blockIdx.x * 128;
  v8f c[2][4];
  gemm_core<KQKV>(A, Bt, m0, n0, As, Bs, c);

  const int lane = threadIdx.x & 31;
  const int wave = threadIdx.x >> 5;
  const int wm = wave & 3, wn = wave >> 2;
  const int lrow = lane & 15, lhi = lane >> 4;
  #pragma unroll
  for (int mi = 0; mi < 2; ++mi)
    #pragma unroll
    for (int ni = 0; ni < 4; ++ni)
      #pragma unroll
      for (int r = 0; r < 8; ++r) {
        const int m = m0 + wm * 32 + mi * 16 + lhi * 8 + r;
        const int n = n0 + wn * 64 + ni * 16 + lrow;
        const int b = m >> 10, y = (m >> 5) & 31, x = m & 31;
        float v = c[mi][ni][r] + biasb[(size_t)b * NQKV + n]
                + rowb[((size_t)b * NY + y) * NQKV + n]
                + colb[((size_t)b * NX + x) * NQKV + n];
        qkv[(size_t)m * NQKV + n] = v;
      }
}

// output GEMM: cells_out[65536,512] = o_f16 @ W_out_cells. grid (4, 512)
__global__ void __launch_bounds__(256)
gemm_out_kernel(const _Float16* __restrict__ A, const _Float16* __restrict__ Bt,
                float* __restrict__ out) {
  __shared__ __attribute__((aligned(16))) _Float16 As[128 * LDSS];
  __shared__ __attribute__((aligned(16))) _Float16 Bs[128 * LDSS];
  const int m0 = blockIdx.y * 128, n0 = blockIdx.x * 128;
  v8f c[2][4];
  gemm_core<NVDIM>(A, Bt, m0, n0, As, Bs, c);

  const int lane = threadIdx.x & 31;
  const int wave = threadIdx.x >> 5;
  const int wm = wave & 3, wn = wave >> 2;
  const int lrow = lane & 15, lhi = lane >> 4;
  #pragma unroll
  for (int mi = 0; mi < 2; ++mi)
    #pragma unroll
    for (int ni = 0; ni < 4; ++ni)
      #pragma unroll
      for (int r = 0; r < 8; ++r) {
        const int m = m0 + wm * 32 + mi * 16 + lhi * 8 + r;
        const int n = n0 + wn * 64 + ni * 16 + lrow;
        out[(size_t)m * NVDIM + n] = c[mi][ni][r];
      }
}

// ======================= attention =======================
// dir=0: attend along X (fixed y=s, pos from xpos); dir=1: along Y (fixed x=s, ypos).
// One block per (b, s, head). dir=1 also finalizes o = (o_x + o_y)*valid and emits f16.
__global__ void __launch_bounds__(256)
attn_kernel(const float* __restrict__ qkv, const float* __restrict__ pos,
            const float* __restrict__ freqs,
            const unsigned char* __restrict__ rmsk,
            const unsigned char* __restrict__ cmsk,
            const unsigned char* __restrict__ mask,
            float* __restrict__ o, _Float16* __restrict__ o_h, int dir) {
  const int n = blockIdx.x & 7;
  const int s = (blockIdx.x >> 3) & 31;
  const int b = blockIdx.x >> 8;
  const int tid = threadIdx.x;

  __shared__ float qr[32][64];
  __shared__ float kr[32][64];
  __shared__ float vv[32][64];
  __shared__ float sc[32][32];
  __shared__ float validf[32];

  // RoPE q/k + load v
  for (int i = tid; i < 32 * 32; i += 256) {
    const int t = i >> 5, h = i & 31;
    const size_t cell = dir ? ((size_t)(b * NY + t) * NX + s)
                            : ((size_t)(b * NY + s) * NX + t);
    const float* base = qkv + cell * NQKV;
    const float p0 = pos[((size_t)b * 32 + t) * 2 + 0];
    const float p1 = pos[((size_t)b * 32 + t) * 2 + 1];
    const float phi = p0 * freqs[(n * 32 + h) * 2 + 0] + p1 * freqs[(n * 32 + h) * 2 + 1];
    const float cph = cosf(phi), sph = sinf(phi);
    const float q0 = base[n * 64 + 2 * h],       q1 = base[n * 64 + 2 * h + 1];
    const float k0 = base[512 + n * 64 + 2 * h], k1 = base[512 + n * 64 + 2 * h + 1];
    qr[t][2 * h] = q0 * cph - q1 * sph;  qr[t][2 * h + 1] = q0 * sph + q1 * cph;
    kr[t][2 * h] = k0 * cph - k1 * sph;  kr[t][2 * h + 1] = k0 * sph + k1 * cph;
  }
  for (int i = tid; i < 32 * 64; i += 256) {
    const int t = i >> 6, d = i & 63;
    const size_t cell = dir ? ((size_t)(b * NY + t) * NX + s)
                            : ((size_t)(b * NY + s) * NX + t);
    vv[t][d] = qkv[cell * NQKV + 1024 + n * 64 + d];
  }
  if (tid < 32) {
    const int t = tid;
    const int y = dir ? t : s, x = dir ? s : t;
    const bool v = mask[((size_t)b * NY + y) * NX + x] && rmsk[b * NY + y] && cmsk[b * NX + x];
    validf[t] = v ? 1.f : 0.f;
  }
  __syncthreads();

  // scores (scaled, key-masked)
  for (int i = tid; i < 1024; i += 256) {
    const int qi = i >> 5, ki = i & 31;
    float acc = 0.f;
    #pragma unroll 8
    for (int d = 0; d < 64; ++d) acc += qr[qi][d] * kr[ki][d];
    sc[qi][ki] = (validf[ki] > 0.f) ? acc * 0.125f : -1e9f;
  }
  __syncthreads();

  // softmax per query row
  if (tid < 32) {
    float mx = -3.0e38f;
    for (int ki = 0; ki < 32; ++ki) mx = fmaxf(mx, sc[tid][ki]);
    float ssum = 0.f;
    for (int ki = 0; ki < 32; ++ki) { float e = __expf(sc[tid][ki] - mx); sc[tid][ki] = e; ssum += e; }
    const float inv = 1.f / ssum;
    for (int ki = 0; ki < 32; ++ki) sc[tid][ki] *= inv;
  }
  __syncthreads();

  // out = attn @ V
  for (int i = tid; i < 32 * 64; i += 256) {
    const int qi = i >> 6, d = i & 63;
    float acc = 0.f;
    #pragma unroll 8
    for (int ki = 0; ki < 32; ++ki) acc += sc[qi][ki] * vv[ki][d];
    const size_t cell = dir ? ((size_t)(b * NY + qi) * NX + s)
                            : ((size_t)(b * NY + s) * NX + qi);
    const size_t oidx = cell * NVDIM + n * 64 + d;
    if (dir == 0) {
      o[oidx] = acc;
    } else {
      const float v = (o[oidx] + acc) * validf[qi];
      o[oidx] = v;
      o_h[oidx] = (_Float16)v;
    }
  }
}

// ======================= pooling + small projections =======================
__global__ void pool_rows_kernel(const float* __restrict__ o,
                                 const unsigned char* __restrict__ rmsk,
                                 const unsigned char* __restrict__ cmsk,
                                 const unsigned char* __restrict__ mask,
                                 float* __restrict__ rowmean) {
  const int idx = blockIdx.x * 256 + threadIdx.x;   // B*Y*512
  const int d = idx & 511, by = idx >> 9;
  const int b = by >> 5, y = by & 31;
  float ssum = 0.f, cnt = 0.f;
  for (int x = 0; x < NX; ++x) {
    const bool v = mask[((size_t)b * NY + y) * NX + x] && rmsk[b * NY + y] && cmsk[b * NX + x];
    cnt += v ? 1.f : 0.f;
    ssum += o[(((size_t)by) * NX + x) * NVDIM + d];
  }
  rowmean[idx] = ssum / fmaxf(cnt, 1.f);
}

__global__ void pool_cols_kernel(const float* __restrict__ o,
                                 const unsigned char* __restrict__ rmsk,
                                 const unsigned char* __restrict__ cmsk,
                                 const unsigned char* __restrict__ mask,
                                 float* __restrict__ colmean) {
  const int idx = blockIdx.x * 256 + threadIdx.x;   // B*X*512
  const int d = idx & 511, bx = idx >> 9;
  const int b = bx >> 5, x = bx & 31;
  float ssum = 0.f, cnt = 0.f;
  for (int y = 0; y < NY; ++y) {
    const bool v = mask[((size_t)b * NY + y) * NX + x] && rmsk[b * NY + y] && cmsk[b * NX + x];
    cnt += v ? 1.f : 0.f;
    ssum += o[(((size_t)b * NY + y) * NX + x) * NVDIM + d];
  }
  colmean[idx] = ssum / fmaxf(cnt, 1.f);
}

__global__ void pool_glob_kernel(const float* __restrict__ o,
                                 const unsigned char* __restrict__ rmsk,
                                 const unsigned char* __restrict__ cmsk,
                                 const unsigned char* __restrict__ mask,
                                 float* __restrict__ gmean) {
  const int idx = blockIdx.x * 256 + threadIdx.x;   // B*512
  const int d = idx & 511, b = idx >> 9;
  float ssum = 0.f, cnt = 0.f;
  for (int p = 0; p < NY * NX; ++p) {
    const int y = p >> 5, x = p & 31;
    const bool v = mask[((size_t)b * NY + y) * NX + x] && rmsk[b * NY + y] && cmsk[b * NX + x];
    cnt += v ? 1.f : 0.f;
    ssum += o[((size_t)b * NY * NX + p) * NVDIM + d];
  }
  gmean[idx] = ssum / fmaxf(cnt, 1.f);
}

// out[m, 0:256] = pin[m, 0:512] @ W[512,256]; one block per m
__global__ void proj512_kernel(const float* __restrict__ pin,
                               const float* __restrict__ W,
                               float* __restrict__ outp) {
  const int m = blockIdx.x;
  __shared__ float pr[512];
  for (int i = threadIdx.x; i < 512; i += 256) pr[i] = pin[(size_t)m * 512 + i];
  __syncthreads();
  float acc = 0.f;
  for (int d = 0; d < 512; ++d) acc += pr[d] * W[(size_t)d * 256 + threadIdx.x];
  outp[(size_t)m * 256 + threadIdx.x] = acc;
}

// ======================= launch =======================
extern "C" void kernel_launch(void* const* d_in, const int* in_sizes, int n_in,
                              void* d_out, int out_size, void* d_ws, size_t ws_size,
                              hipStream_t stream) {
  (void)in_sizes; (void)n_in; (void)out_size; (void)ws_size;
  const float* cells      = (const float*)d_in[0];
  const float* rows       = (const float*)d_in[1];
  const float* cols       = (const float*)d_in[2];
  const float* globl      = (const float*)d_in[3];
  const float* fcells     = (const float*)d_in[4];
  const float* flavours   = (const float*)d_in[5];
  const float* ypos       = (const float*)d_in[6];
  const float* xpos       = (const float*)d_in[7];
  const unsigned char* rmsk = (const unsigned char*)d_in[8];
  const unsigned char* cmsk = (const unsigned char*)d_in[9];
  const unsigned char* mask = (const unsigned char*)d_in[10];
  const float* freqs      = (const float*)d_in[11];
  const float* W_cells    = (const float*)d_in[12];
  const float* W_rows     = (const float*)d_in[13];
  const float* W_cols     = (const float*)d_in[14];
  const float* W_globl    = (const float*)d_in[15];
  const float* W_fcells   = (const float*)d_in[16];
  const float* W_flavours = (const float*)d_in[17];
  const float* W_out_cells= (const float*)d_in[18];
  const float* W_out_rows = (const float*)d_in[19];
  const float* W_out_cols = (const float*)d_in[20];
  const float* W_out_globl= (const float*)d_in[21];
  float* out = (float*)d_out;

  char* ws = (char*)d_ws;
  size_t off = 0;
  auto alloc = [&](size_t bytes) -> void* {
    off = (off + 255) & ~(size_t)255;
    void* p = ws + off;
    off += bytes;
    return p;
  };
  _Float16* Aq     = (_Float16*)alloc((size_t)MCELL * KQKV * 2);
  _Float16* WqkvT  = (_Float16*)alloc((size_t)NQKV * KQKV * 2);
  _Float16* WoutT  = (_Float16*)alloc((size_t)512 * 512 * 2);
  float* biasb     = (float*)alloc((size_t)NB * NQKV * 4);
  float* rowb      = (float*)alloc((size_t)NB * NY * NQKV * 4);
  float* colb      = (float*)alloc((size_t)NB * NX * NQKV * 4);
  float* qkv       = (float*)alloc((size_t)MCELL * NQKV * 4);
  float* o         = (float*)alloc((size_t)MCELL * NVDIM * 4);
  _Float16* o_h    = (_Float16*)alloc((size_t)MCELL * NVDIM * 2);
  float* rowmean   = (float*)alloc((size_t)NB * NY * NVDIM * 4);
  float* colmean   = (float*)alloc((size_t)NB * NX * NVDIM * 4);
  float* gmean     = (float*)alloc((size_t)NB * NVDIM * 4);

  // ---- prep ----
  prepA_kernel<<<MCELL, 256, 0, stream>>>(cells, fcells, Aq);
  prepWqkvT_kernel<<<NQKV, 256, 0, stream>>>(W_cells, W_fcells, WqkvT);
  prepWoutT_kernel<<<512, 256, 0, stream>>>(W_out_cells, WoutT);
  bias_kernel<<<NB, 256, 0, stream>>>(globl, flavours, W_globl, W_fcells, W_flavours, biasb);
  dense256_kernel<<<dim3(6, NB * NY), 256, 0, stream>>>(rows, W_rows, rowb);
  dense256_kernel<<<dim3(6, NB * NX), 256, 0, stream>>>(cols, W_cols, colb);

  // ---- fused qkv (WMMA) ----
  gemm_qkv_kernel<<<dim3(NQKV / 128, MCELL / 128), 256, 0, stream>>>(
      Aq, WqkvT, biasb, rowb, colb, qkv);

  // ---- axial attention (X then Y; Y pass finalizes o and emits f16) ----
  attn_kernel<<<NB * 32 * 8, 256, 0, stream>>>(qkv, xpos, freqs, rmsk, cmsk, mask, o, o_h, 0);
  attn_kernel<<<NB * 32 * 8, 256, 0, stream>>>(qkv, ypos, freqs, rmsk, cmsk, mask, o, o_h, 1);

  // ---- outputs ----
  gemm_out_kernel<<<dim3(NVDIM / 128, MCELL / 128), 256, 0, stream>>>(o_h, WoutT, out);

  pool_rows_kernel<<<(NB * NY * NVDIM) / 256, 256, 0, stream>>>(o, rmsk, cmsk, mask, rowmean);
  pool_cols_kernel<<<(NB * NX * NVDIM) / 256, 256, 0, stream>>>(o, rmsk, cmsk, mask, colmean);
  pool_glob_kernel<<<(NB * NVDIM) / 256, 256, 0, stream>>>(o, rmsk, cmsk, mask, gmean);

  float* rows_out  = out + (size_t)MCELL * 512;                 // 33,554,432
  float* cols_out  = rows_out + (size_t)NB * NY * 256;          // +524,288
  float* globl_out = cols_out + (size_t)NB * NX * 256;          // +524,288
  proj512_kernel<<<NB * NY, 256, 0, stream>>>(rowmean, W_out_rows, rows_out);
  proj512_kernel<<<NB * NX, 256, 0, stream>>>(colmean, W_out_cols, cols_out);
  proj512_kernel<<<NB, 256, 0, stream>>>(gmean, W_out_globl, globl_out);
}
